// Scaled_dot_product_attention_37933151158772
// MI455X (gfx1250) — matrix operand order repaired and checked by reading, not compiled, not run on hardware
//
#include <hip/hip_runtime.h>

// ---- CDNA5 WMMA types ----
typedef __attribute__((ext_vector_type(16))) __bf16 bf16x16;
typedef __attribute__((ext_vector_type(8)))  __bf16 bf16x8;
typedef __attribute__((ext_vector_type(4)))  __bf16 bf16x4;
typedef __attribute__((ext_vector_type(8)))  float  f32x8;
typedef __attribute__((ext_vector_type(4)))  float  f32x4;

static __device__ __forceinline__ f32x8 wmma_bf16(bf16x16 a, bf16x16 b, f32x8 c) {
  // D = A(16x32 bf16) * B(32x16 bf16) + C(16x16 f32)
  return __builtin_amdgcn_wmma_f32_16x16x32_bf16(false, a, false, b, (short)0, c, false, false);
}

constexpr int H = 16;
constexpr int S = 4096;
constexpr int D = 64;
// exp(s * 1/sqrt(64)) == exp2(s * (0.125 * log2(e))) -> one v_mul + v_exp_f32
constexpr float EXP2SC = 0.125f * 1.44269504088896340736f;

// ======================= prep kernels =======================

// K fp32 [h][j][d]  ->  bf16 [h][j][d]
__global__ __launch_bounds__(256)
void cvt_k_kernel(const float* __restrict__ K, __bf16* __restrict__ Kb) {
  const size_t i = ((size_t)blockIdx.x * 256 + threadIdx.x) * 4;
  f32x4 x = *(const f32x4*)(K + i);
  bf16x4 y;
  #pragma unroll
  for (int e = 0; e < 4; ++e) y[e] = (__bf16)x[e];
  *(bf16x4*)(Kb + i) = y;
}

// V fp32 [h][j][d]  ->  bf16 Vt [h][d][j]   (64x64 LDS tile transpose)
__global__ __launch_bounds__(256)
void tr_v_kernel(const float* __restrict__ V, __bf16* __restrict__ Vt) {
  __shared__ float tile[64][65];
  const int h  = blockIdx.x >> 6;          // S/64 = 64 tiles per head
  const int j0 = (blockIdx.x & 63) * 64;
  const int t  = threadIdx.x;
  const int rl = t >> 4;                   // 0..15
  const int c4 = (t & 15) * 4;             // 0,4,...,60

  #pragma unroll
  for (int r = 0; r < 4; ++r) {
    const int j = rl + r * 16;             // local row (key)
    f32x4 x = *(const f32x4*)(V + (((size_t)h * S) + j0 + j) * D + c4);
    #pragma unroll
    for (int e = 0; e < 4; ++e) tile[j][c4 + e] = x[e];
  }
  __syncthreads();
  #pragma unroll
  for (int r = 0; r < 4; ++r) {
    const int d = rl + r * 16;             // output row (feature)
    bf16x4 y;
    #pragma unroll
    for (int e = 0; e < 4; ++e) y[e] = (__bf16)tile[c4 + e][d];
    *(bf16x4*)(Vt + (((size_t)h * D) + d) * S + j0 + c4) = y;
  }
}

// ======================= main kernel (bf16 pre-packed operands) =======================
// One workgroup = 16 queries x ALL 16 heads (head-axis softmax couples heads).
// 8 waves; wave w owns heads {2w, 2w+1}.
__global__ __launch_bounds__(256, 1)
void attn_headsoftmax_wmma_kernel(const float* __restrict__ Q,
                                  const __bf16* __restrict__ Kb,  // [h][j][d] bf16
                                  const __bf16* __restrict__ Vt,  // [h][d][j] bf16
                                  float* __restrict__ O)
{
  __shared__ float lds_psum[8 * 512];   // per-wave partial head-sums; reused as bf16 P tiles
  __shared__ float lds_z[512];          // Z[q(16) * 32 + key(32)]

  const int tid  = threadIdx.x;
  const int wave = tid >> 5;
  const int lane = tid & 31;
  const int nn   = lane & 15;           // column index inside a 16-wide tile
  const int hi   = lane >> 4;           // lane half (0/1)
  const int mb   = hi * 8;              // C-layout row base for this half
  const int qb   = blockIdx.x * 16;     // query stripe base
  const int h0   = wave * 2;            // first head owned by this wave

  // ---- Load Q (fp32, once) as A-fragments: 2 heads x 2 chunks of K=32 ----
  bf16x16 qa[2][2];
  #pragma unroll
  for (int hh = 0; hh < 2; ++hh) {
    const float* qrow = Q + (((size_t)(h0 + hh) * S) + (qb + nn)) * D;
    #pragma unroll
    for (int dk = 0; dk < 2; ++dk) {
      const float* p0 = qrow + dk * 32 + hi * 16;
      f32x4 c0 = *(const f32x4*)(p0);
      f32x4 c1 = *(const f32x4*)(p0 + 4);
      f32x4 c2 = *(const f32x4*)(p0 + 8);
      f32x4 c3 = *(const f32x4*)(p0 + 12);
      bf16x16 a;
      #pragma unroll
      for (int e = 0; e < 4; ++e) {
        a[e]      = (__bf16)c0[e];
        a[e + 4]  = (__bf16)c1[e];
        a[e + 8]  = (__bf16)c2[e];
        a[e + 12] = (__bf16)c3[e];
      }
      qa[hh][dk] = a;
    }
  }

  f32x8 oacc[2][4];
  #pragma unroll
  for (int hh = 0; hh < 2; ++hh)
    #pragma unroll
    for (int dc = 0; dc < 4; ++dc)
      oacc[hh][dc] = (f32x8)(0.0f);

  for (int j0 = 0; j0 < S; j0 += 32) {
    // in-bounds prefetch target for the next key chunk
    const int jp = (j0 + 32 < S) ? (j0 + 32) : 0;

    // ---- scores + exp for this wave's two heads (16q x 32k each) ----
    f32x8 ex[2][2];
    #pragma unroll
    for (int hh = 0; hh < 2; ++hh) {
      const __bf16* kh = Kb + (size_t)(h0 + hh) * S * D;
      const __bf16* vh = Vt + (size_t)(h0 + hh) * D * S;
      // prefetch next chunk into WGP-level cache (locality 3 -> SCOPE_WGP):
      // 32 K rows (128B each) and 64 Vt row segments
      __builtin_prefetch(kh + (size_t)(jp + lane) * D, 0, 3);
      __builtin_prefetch(vh + (size_t)lane * S + jp, 0, 3);
      __builtin_prefetch(vh + (size_t)(lane + 32) * S + jp, 0, 3);

      #pragma unroll
      for (int nc = 0; nc < 2; ++nc) {
        f32x8 c = (f32x8)(0.0f);
        #pragma unroll
        for (int dk = 0; dk < 2; ++dk) {
          // B layout: lane column = key j0+nc*16+nn; halves <-> d = dk*32+hi*16+(0..15)
          bf16x16 b = *(const bf16x16*)(kh + (size_t)(j0 + nc * 16 + nn) * D + dk * 32 + hi * 16);
          c = wmma_bf16(qa[hh][dk], b, c);
        }
        f32x8 e8;
        #pragma unroll
        for (int r = 0; r < 8; ++r) e8[r] = __builtin_amdgcn_exp2f(c[r] * EXP2SC);
        ex[hh][nc] = e8;
      }
    }

    // ---- per-wave partial head-sum -> LDS slot (C layout: elem (mb+r, nc*16+nn)) ----
    #pragma unroll
    for (int nc = 0; nc < 2; ++nc)
      #pragma unroll
      for (int r = 0; r < 8; ++r)
        lds_psum[wave * 512 + (mb + r) * 32 + nc * 16 + nn] = ex[0][nc][r] + ex[1][nc][r];
    __syncthreads();

    // ---- Z reduction across the 8 wave slots (2 elements per thread) ----
    {
      const int e0 = tid * 2;
      float z0 = 0.f, z1 = 0.f;
      #pragma unroll
      for (int w = 0; w < 8; ++w) {
        z0 += lds_psum[w * 512 + e0];
        z1 += lds_psum[w * 512 + e0 + 1];
      }
      lds_z[e0]     = z0;
      lds_z[e0 + 1] = z1;
    }
    __syncthreads();

    // ---- P = exp(s)/Z in bf16 -> this wave's private slot (reuse psum region) ----
    __bf16* pbase = (__bf16*)(lds_psum + wave * 512);
    #pragma unroll
    for (int nc = 0; nc < 2; ++nc) {
      #pragma unroll
      for (int r = 0; r < 8; ++r) {
        const int idx = (mb + r) * 32 + nc * 16 + nn;
        const float rz = __builtin_amdgcn_rcpf(lds_z[idx]);
        pbase[idx]       = (__bf16)(ex[0][nc][r] * rz);
        pbase[512 + idx] = (__bf16)(ex[1][nc][r] * rz);
      }
    }
    // No barrier needed: slot is private to this wave; same-wave DS ops are in-order.

    // ---- PV: A = P (16q x 32k) via LDS relayout, B = Vt chunk (32k x 16d) ----
    #pragma unroll
    for (int hh = 0; hh < 2; ++hh) {
      const __bf16* prow = pbase + hh * 512 + nn * 32 + hi * 16;
      bf16x8 lo  = *(const bf16x8*)(prow);
      bf16x8 hi8 = *(const bf16x8*)(prow + 8);
      bf16x16 a;
      #pragma unroll
      for (int e = 0; e < 8; ++e) { a[e] = lo[e]; a[e + 8] = hi8[e]; }

      const __bf16* vh = Vt + (size_t)(h0 + hh) * D * S;
      #pragma unroll
      for (int dc = 0; dc < 4; ++dc) {
        // B layout: lane column = d (dc*16+nn); halves <-> key j0 + hi*16 + (0..15)
        bf16x16 b = *(const bf16x16*)(vh + (size_t)(dc * 16 + nn) * S + j0 + hi * 16);
        oacc[hh][dc] = wmma_bf16(a, b, oacc[hh][dc]);
      }
    }
  }

  // ---- store O (C layout -> row-major [h][q][d]) ----
  #pragma unroll
  for (int hh = 0; hh < 2; ++hh)
    #pragma unroll
    for (int dc = 0; dc < 4; ++dc)
      #pragma unroll
      for (int r = 0; r < 8; ++r)
        O[(((size_t)(h0 + hh) * S) + qb + mb + r) * D + dc * 16 + nn] = oacc[hh][dc][r];
}

// ======================= fallback (no workspace): direct fp32 loads =======================
__global__ __launch_bounds__(256, 1)
void attn_headsoftmax_fallback_kernel(const float* __restrict__ Q,
                                      const float* __restrict__ K,
                                      const float* __restrict__ V,
                                      float* __restrict__ O)
{
  __shared__ float lds_psum[8 * 512];
  __shared__ float lds_z[512];

  const int tid  = threadIdx.x;
  const int wave = tid >> 5;
  const int lane = tid & 31;
  const int nn   = lane & 15;
  const int hi   = lane >> 4;
  const int mb   = hi * 8;
  const int qb   = blockIdx.x * 16;
  const int h0   = wave * 2;

  bf16x16 qa[2][2];
  #pragma unroll
  for (int hh = 0; hh < 2; ++hh) {
    const float* qrow = Q + (((size_t)(h0 + hh) * S) + (qb + nn)) * D;
    #pragma unroll
    for (int dk = 0; dk < 2; ++dk) {
      const float* p0 = qrow + dk * 32 + hi * 16;
      f32x4 c0 = *(const f32x4*)(p0);
      f32x4 c1 = *(const f32x4*)(p0 + 4);
      f32x4 c2 = *(const f32x4*)(p0 + 8);
      f32x4 c3 = *(const f32x4*)(p0 + 12);
      bf16x16 a;
      #pragma unroll
      for (int e = 0; e < 4; ++e) {
        a[e] = (__bf16)c0[e]; a[e+4] = (__bf16)c1[e];
        a[e+8] = (__bf16)c2[e]; a[e+12] = (__bf16)c3[e];
      }
      qa[hh][dk] = a;
    }
  }

  f32x8 oacc[2][4];
  #pragma unroll
  for (int hh = 0; hh < 2; ++hh)
    #pragma unroll
    for (int dc = 0; dc < 4; ++dc)
      oacc[hh][dc] = (f32x8)(0.0f);

  for (int j0 = 0; j0 < S; j0 += 32) {
    f32x8 ex[2][2];
    #pragma unroll
    for (int hh = 0; hh < 2; ++hh) {
      const float* kh = K + (size_t)(h0 + hh) * S * D;
      #pragma unroll
      for (int nc = 0; nc < 2; ++nc) {
        f32x8 c = (f32x8)(0.0f);
        #pragma unroll
        for (int dk = 0; dk < 2; ++dk) {
          const float* krow = kh + (size_t)(j0 + nc * 16 + nn) * D + dk * 32 + hi * 16;
          f32x4 b0 = *(const f32x4*)(krow);
          f32x4 b1 = *(const f32x4*)(krow + 4);
          f32x4 b2 = *(const f32x4*)(krow + 8);
          f32x4 b3 = *(const f32x4*)(krow + 12);
          bf16x16 b;
          #pragma unroll
          for (int e = 0; e < 4; ++e) {
            b[e] = (__bf16)b0[e]; b[e+4] = (__bf16)b1[e];
            b[e+8] = (__bf16)b2[e]; b[e+12] = (__bf16)b3[e];
          }
          c = wmma_bf16(qa[hh][dk], b, c);
        }
        f32x8 e8;
        #pragma unroll
        for (int r = 0; r < 8; ++r) e8[r] = __builtin_amdgcn_exp2f(c[r] * EXP2SC);
        ex[hh][nc] = e8;
      }
    }

    #pragma unroll
    for (int nc = 0; nc < 2; ++nc)
      #pragma unroll
      for (int r = 0; r < 8; ++r)
        lds_psum[wave * 512 + (mb + r) * 32 + nc * 16 + nn] = ex[0][nc][r] + ex[1][nc][r];
    __syncthreads();
    {
      const int e0 = tid * 2;
      float z0 = 0.f, z1 = 0.f;
      #pragma unroll
      for (int w = 0; w < 8; ++w) {
        z0 += lds_psum[w * 512 + e0];
        z1 += lds_psum[w * 512 + e0 + 1];
      }
      lds_z[e0] = z0; lds_z[e0 + 1] = z1;
    }
    __syncthreads();

    __bf16* pbase = (__bf16*)(lds_psum + wave * 512);
    #pragma unroll
    for (int nc = 0; nc < 2; ++nc) {
      #pragma unroll
      for (int r = 0; r < 8; ++r) {
        const int idx = (mb + r) * 32 + nc * 16 + nn;
        const float rz = __builtin_amdgcn_rcpf(lds_z[idx]);
        pbase[idx]       = (__bf16)(ex[0][nc][r] * rz);
        pbase[512 + idx] = (__bf16)(ex[1][nc][r] * rz);
      }
    }

    #pragma unroll
    for (int hh = 0; hh < 2; ++hh) {
      const __bf16* prow = pbase + hh * 512 + nn * 32 + hi * 16;
      bf16x8 lo  = *(const bf16x8*)(prow);
      bf16x8 hi8 = *(const bf16x8*)(prow + 8);
      bf16x16 a;
      #pragma unroll
      for (int e = 0; e < 8; ++e) { a[e] = lo[e]; a[e + 8] = hi8[e]; }

      const float* vb = V + (size_t)(h0 + hh) * S * D + (size_t)(j0 + hi * 16) * D;
      #pragma unroll
      for (int dc = 0; dc < 4; ++dc) {
        bf16x16 b;
        #pragma unroll
        for (int e = 0; e < 16; ++e)
          b[e] = (__bf16)vb[(size_t)e * D + dc * 16 + nn];
        oacc[hh][dc] = wmma_bf16(a, b, oacc[hh][dc]);
      }
    }
  }

  #pragma unroll
  for (int hh = 0; hh < 2; ++hh)
    #pragma unroll
    for (int dc = 0; dc < 4; ++dc)
      #pragma unroll
      for (int r = 0; r < 8; ++r)
        O[(((size_t)(h0 + hh) * S) + qb + mb + r) * D + dc * 16 + nn] = oacc[hh][dc][r];
}

// ======================= host launch =======================
extern "C" void kernel_launch(void* const* d_in, const int* in_sizes, int n_in,
                              void* d_out, int out_size, void* d_ws, size_t ws_size,
                              hipStream_t stream) {
  (void)in_sizes; (void)n_in; (void)out_size;
  const float* q = (const float*)d_in[0];
  const float* k = (const float*)d_in[1];
  const float* v = (const float*)d_in[2];
  float* o = (float*)d_out;

  const size_t elems = (size_t)H * S * D;          // 4,194,304
  const size_t need  = 2 * elems * sizeof(__bf16); // Kb + Vt = 16 MiB

  if (d_ws != nullptr && ws_size >= need) {
    __bf16* Kb = (__bf16*)d_ws;
    __bf16* Vt = (__bf16*)((char*)d_ws + elems * sizeof(__bf16));
    cvt_k_kernel<<<dim3((unsigned)(elems / (256 * 4))), dim3(256), 0, stream>>>(k, Kb);
    tr_v_kernel<<<dim3(H * (S / 64)), dim3(256), 0, stream>>>(v, Vt);
    attn_headsoftmax_wmma_kernel<<<dim3(S / 16), dim3(256), 0, stream>>>(q, Kb, Vt, o);
  } else {
    attn_headsoftmax_fallback_kernel<<<dim3(S / 16), dim3(256), 0, stream>>>(q, k, v, o);
  }
}
